// MultiHeadAttentionWithRoPE_87711822119533
// MI455X (gfx1250) — matrix operand order repaired and checked by reading, not compiled, not run on hardware
//
#include <hip/hip_runtime.h>
#include <hip/hip_bf16.h>

// ---------------------------------------------------------------------------
// MHA + RoPE for MI455X (gfx1250, wave32, WMMA 16x16x32 f16 / f32-acc)
// B=2, S=2048, D=1024, H=16, DH=64
// Async global->LDS staging (ASYNCcnt) + double-buffered WMMA GEMM.
// ---------------------------------------------------------------------------

typedef _Float16 v16h __attribute__((ext_vector_type(16)));
typedef _Float16 v8h  __attribute__((ext_vector_type(8)));
typedef float    v8f  __attribute__((ext_vector_type(8)));
typedef int      v4i  __attribute__((ext_vector_type(4)));

#define WMMA_F16(a, b, c) \
    __builtin_amdgcn_wmma_f32_16x16x32_f16(false, (a), false, (b), (short)0, (c), false, false)

// ---- CDNA5 async global->LDS copy (16B per lane) ---------------------------
// Probe-revealed signature: param1 = AS1 v4i* (global src), param2 = LDS dst.
#if __has_builtin(__builtin_amdgcn_global_load_async_to_lds_b128)
__device__ __forceinline__ void async_copy16(const _Float16* gp, _Float16* lp) {
    __builtin_amdgcn_global_load_async_to_lds_b128(
        (__attribute__((address_space(1))) v4i*)(uintptr_t)gp,
        (__attribute__((address_space(3))) v4i*)(uintptr_t)lp, 0, 0);
}
#else
__device__ __forceinline__ void async_copy16(const _Float16* gp, _Float16* lp) {
    asm volatile("global_load_async_to_lds_b128 %0, %1, off"
                 :: "v"((unsigned)(uintptr_t)lp),
                    "v"((unsigned long long)(uintptr_t)gp)
                 : "memory");
}
#endif

#if __has_builtin(__builtin_amdgcn_s_wait_asynccnt)
#define WAIT_ASYNC() __builtin_amdgcn_s_wait_asynccnt(0)
#else
#define WAIT_ASYNC() asm volatile("s_wait_asynccnt 0x0" ::: "memory")
#endif

// ---- fragment loaders (ISA 7.12.2 layouts, wave32) -------------------------
// A (16x32 f16): lane L holds row M=(L&15); halves j=0..7 -> K = k0+j,
// j=8..15 -> K = k0+16+(j-8), with k0 = 8*(L>>4).
__device__ __forceinline__ v16h load_frag_a(const _Float16* base, int stride,
                                            int row0, int ks, int lane) {
    int m  = row0 + (lane & 15);
    int k0 = ks + 8 * (lane >> 4);
    const _Float16* p = base + m * stride + k0;
    union { v16h v; v8h h[2]; } u;
    u.h[0] = *(const v8h*)(p);
    u.h[1] = *(const v8h*)(p + 16);
    return u.v;
}
// B (32x16 f16) from B^T stored row-major [N][K]: lane L holds col N=(L&15);
// halves j=0..7 -> K = k0+j, j=8..15 -> K = k0+16+(j-8), k0 = ks + 8*(L>>4).
__device__ __forceinline__ v16h load_frag_b(const _Float16* base, int stride,
                                            int n0, int ks, int lane) {
    int n  = n0 + (lane & 15);
    int k0 = ks + 8 * (lane >> 4);
    const _Float16* p = base + n * stride + k0;
    union { v16h v; v8h h[2]; } u;
    u.h[0] = *(const v8h*)(p);
    u.h[1] = *(const v8h*)(p + 16);
    return u.v;
}

// ---------------------------------------------------------------------------
// f32 -> f16 flat convert (4 elems / thread)
// ---------------------------------------------------------------------------
__global__ __launch_bounds__(256) void cvt_f32_to_f16(const float* __restrict__ in,
                                                      _Float16* __restrict__ out, int n) {
    int i = (blockIdx.x * 256 + threadIdx.x) * 4;
    if (i + 3 < n) {
        float4 v = *(const float4*)(in + i);
        out[i + 0] = (_Float16)v.x;
        out[i + 1] = (_Float16)v.y;
        out[i + 2] = (_Float16)v.z;
        out[i + 3] = (_Float16)v.w;
    }
}

// ---------------------------------------------------------------------------
// transpose + convert: w[K][N] f32 -> wT[N][K] f16 (32x32 LDS tiles)
// ---------------------------------------------------------------------------
__global__ __launch_bounds__(256) void transpose_cvt(const float* __restrict__ w,
                                                     _Float16* __restrict__ wT,
                                                     int K, int N) {
    __shared__ float t[32][33];
    int bx = blockIdx.x * 32, by = blockIdx.y * 32;
    int tx = threadIdx.x, ty = threadIdx.y;
    for (int j = 0; j < 32; j += 8)
        t[ty + j][tx] = w[(size_t)(by + ty + j) * N + bx + tx];
    __syncthreads();
    for (int j = 0; j < 32; j += 8)
        wT[(size_t)(bx + ty + j) * K + by + tx] = (_Float16)t[tx][ty + j];
}

// ---------------------------------------------------------------------------
// WMMA GEMM: out[M][N] = A[M][K](f16) * BT[N][K](f16)^T + bias
// block = 256 thr (8 waves), tile 128(M) x 64(N), K-step 64.
// Double-buffered LDS, async global->LDS staging overlapped with WMMA.
// ---------------------------------------------------------------------------
#define G_TM 128
#define G_TN 64
#define G_TK 64
#define G_LS 72   // LDS half-stride (16B-aligned chunks, bank-conflict pad)

__global__ __launch_bounds__(256) void gemm_wmma_f16(
    const _Float16* __restrict__ A, const _Float16* __restrict__ BT,
    const float* __restrict__ bias, _Float16* __restrict__ outH,
    float* __restrict__ outF, int M, int N, int K) {
    __shared__ __align__(16) _Float16 As[2][G_TM * G_LS];
    __shared__ __align__(16) _Float16 Bs[2][G_TN * G_LS];

    int tid  = threadIdx.x;
    int lane = tid & 31;
    int wave = tid >> 5;
    int waveM = (wave & 3) * 32;
    int waveN = (wave >> 2) * 32;
    int blockM = blockIdx.x * G_TM;
    int blockN = blockIdx.y * G_TN;

    v8f acc00 = {}, acc01 = {}, acc10 = {}, acc11 = {};

    auto stage = [&](int kb, int buf) {
#pragma unroll
        for (int i = 0; i < 4; i++) {          // A: 1024 x 16B chunks
            int idx = tid + i * 256;
            int r = idx >> 3, c = (idx & 7) * 8;
            async_copy16(&A[(size_t)(blockM + r) * K + kb + c],
                         &As[buf][r * G_LS + c]);
        }
#pragma unroll
        for (int i = 0; i < 2; i++) {          // B^T: 512 x 16B chunks
            int idx = tid + i * 256;
            int r = idx >> 3, c = (idx & 7) * 8;
            async_copy16(&BT[(size_t)(blockN + r) * K + kb + c],
                         &Bs[buf][r * G_LS + c]);
        }
    };

    stage(0, 0);
    int buf = 0;
    for (int kb = 0; kb < K; kb += G_TK) {
        WAIT_ASYNC();                 // my async writes to LDS complete
        __syncthreads();              // everyone's writes visible
        if (kb + G_TK < K) stage(kb + G_TK, buf ^ 1);   // prefetch next tile

        const _Float16* as = As[buf];
        const _Float16* bs = Bs[buf];
#pragma unroll
        for (int ks = 0; ks < G_TK; ks += 32) {
            v16h a0 = load_frag_a(as, G_LS, waveM + 0,  ks, lane);
            v16h a1 = load_frag_a(as, G_LS, waveM + 16, ks, lane);
            v16h b0 = load_frag_b(bs, G_LS, waveN + 0,  ks, lane);
            v16h b1 = load_frag_b(bs, G_LS, waveN + 16, ks, lane);
            acc00 = WMMA_F16(a0, b0, acc00);
            acc01 = WMMA_F16(a0, b1, acc01);
            acc10 = WMMA_F16(a1, b0, acc10);
            acc11 = WMMA_F16(a1, b1, acc11);
        }
        buf ^= 1;
    }

    // epilogue: C/D layout -> M = r + 8*(lane>=16), N = lane&15
    int rowoff = (lane >> 4) * 8;
    v8f* accs[2][2] = {{&acc00, &acc01}, {&acc10, &acc11}};
#pragma unroll
    for (int i = 0; i < 2; i++) {
#pragma unroll
        for (int j = 0; j < 2; j++) {
            int n = blockN + waveN + j * 16 + (lane & 15);
            float bv = bias[n];
#pragma unroll
            for (int r = 0; r < 8; r++) {
                int m = blockM + waveM + i * 16 + r + rowoff;
                float val = (*accs[i][j])[r] + bv;
                if (outF) outF[(size_t)m * N + n] = val;
                else      outH[(size_t)m * N + n] = (_Float16)val;
            }
        }
    }
}

// ---------------------------------------------------------------------------
// RoPE in-place on f16 [B,S,D] with D = H*64; reference semantics:
// repeat(cos,2) paired with half-split rotate.
// ---------------------------------------------------------------------------
__global__ __launch_bounds__(256) void rope_kernel(_Float16* __restrict__ T,
                                                   int B, int S, int H, int D) {
    int idx = blockIdx.x * 256 + threadIdx.x;           // B*S*H*32 threads
    int i = idx & 31;
    int h = (idx >> 5) % H;
    int rem = idx / (32 * H);
    int s = rem % S;
    int b = rem / S;
    if (b >= B) return;

    size_t base = (size_t)(b * S + s) * D + h * 64;
    float q1 = (float)T[base + i];
    float q2 = (float)T[base + i + 32];

    const float NLOG = -9.2103403719761836f / 32.0f;    // -ln(10000)/32
    float fi1 = (float)(i >> 1);
    float fi2 = fi1 + 16.0f;
    float a1 = (float)s * __expf(fi1 * NLOG);
    float a2 = (float)s * __expf(fi2 * NLOG);
    float s1, c1, s2, c2;
    sincosf(a1, &s1, &c1);
    sincosf(a2, &s2, &c2);

    T[base + i]      = (_Float16)(q1 * c1 - q2 * s1);
    T[base + i + 32] = (_Float16)(q2 * c2 + q1 * s2);
}

// ---------------------------------------------------------------------------
// Flash attention: block = (b, h, 128 Q rows); 8 waves x 16 rows.
// K tile async-staged; V staged with transpose; online softmax; WMMA x16.
// ---------------------------------------------------------------------------
#define A_QT 128
#define A_KT 64
#define A_LS 72

__global__ __launch_bounds__(256) void attn_wmma(
    const _Float16* __restrict__ Q, const _Float16* __restrict__ Kt,
    const _Float16* __restrict__ V, const int* __restrict__ mask,
    _Float16* __restrict__ O, int B, int S, int H) {
    __shared__ __align__(16) _Float16 Qs[A_QT * A_LS];
    __shared__ __align__(16) _Float16 Ks[A_KT * A_LS];
    __shared__ __align__(16) _Float16 Vt[A_KT * A_LS];      // [dh][key]
    __shared__ __align__(16) _Float16 Ps[8 * 16 * A_LS];    // per-wave P
    __shared__ float mk[A_KT];

    const int D = 1024, DH = 64;
    int tid  = threadIdx.x;
    int lane = tid & 31;
    int wave = tid >> 5;
    int b  = blockIdx.z;
    int h  = blockIdx.y;
    int q0 = blockIdx.x * A_QT;

    // stage Q tile (128 rows x 64 halves) via async copies
#pragma unroll
    for (int i = 0; i < 4; i++) {
        int idx = tid + i * 256;
        int r = idx >> 3, c = (idx & 7) * 8;
        async_copy16(&Q[(size_t)(b * S + q0 + r) * D + h * DH + c],
                     &Qs[r * A_LS + c]);
    }
    WAIT_ASYNC();
    __syncthreads();

    v16h aq0 = load_frag_a(Qs, A_LS, wave * 16, 0,  lane);
    v16h aq1 = load_frag_a(Qs, A_LS, wave * 16, 32, lane);

    float mrow[8], lrow[8];
#pragma unroll
    for (int r = 0; r < 8; r++) { mrow[r] = -1e30f; lrow[r] = 0.0f; }
    v8f o0 = {}, o1 = {}, o2 = {}, o3 = {};
    v8f* ofr[4] = {&o0, &o1, &o2, &o3};

    _Float16* Pw = &Ps[wave * 16 * A_LS];
    int rowoff = (lane >> 4) * 8;

    for (int kv = 0; kv < S; kv += A_KT) {
        // stage K tile [key][dh] asynchronously
#pragma unroll
        for (int i = 0; i < 2; i++) {
            int idx = tid + i * 256;
            int r = idx >> 3, c = (idx & 7) * 8;
            async_copy16(&Kt[(size_t)(b * S + kv + r) * D + h * DH + c],
                         &Ks[r * A_LS + c]);
        }
        // stage V transposed -> Vt[dh][key] (sync: transposed scatter)
#pragma unroll
        for (int i = 0; i < 2; i++) {
            int idx = tid + i * 256;
            int r = idx >> 3, c = (idx & 7) * 8;
            v8h vv = *(const v8h*)&V[(size_t)(b * S + kv + r) * D + h * DH + c];
#pragma unroll
            for (int e = 0; e < 8; e++) Vt[(c + e) * A_LS + r] = vv[e];
        }
        if (tid < A_KT) mk[tid] = -1e9f * (float)mask[(size_t)b * S + kv + tid];
        WAIT_ASYNC();
        __syncthreads();

        // S = Q K^T  (4 key-subtiles x 2 K-steps = 8 WMMAs)
        v8f sf[4];
#pragma unroll
        for (int j = 0; j < 4; j++) {
            v8f acc = {};
            v16h bk0 = load_frag_b(Ks, A_LS, j * 16, 0,  lane);
            v16h bk1 = load_frag_b(Ks, A_LS, j * 16, 32, lane);
            acc = WMMA_F16(aq0, bk0, acc);
            acc = WMMA_F16(aq1, bk1, acc);
            sf[j] = acc;
        }

        // scale + mask + row max (width-16 shuffle reduction matches C layout)
        float mcol[4];
#pragma unroll
        for (int j = 0; j < 4; j++) mcol[j] = mk[j * 16 + (lane & 15)];
        float tmax[8];
#pragma unroll
        for (int r = 0; r < 8; r++) tmax[r] = -1e30f;
#pragma unroll
        for (int j = 0; j < 4; j++)
#pragma unroll
            for (int r = 0; r < 8; r++) {
                float v = sf[j][r] * 0.125f + mcol[j];
                sf[j][r] = v;
                tmax[r] = fmaxf(tmax[r], v);
            }
#pragma unroll
        for (int off = 1; off < 16; off <<= 1)
#pragma unroll
            for (int r = 0; r < 8; r++)
                tmax[r] = fmaxf(tmax[r], __shfl_xor(tmax[r], off, 16));

        float newm[8], tsum[8], corr[8];
#pragma unroll
        for (int r = 0; r < 8; r++) { newm[r] = fmaxf(mrow[r], tmax[r]); tsum[r] = 0.0f; }
#pragma unroll
        for (int j = 0; j < 4; j++)
#pragma unroll
            for (int r = 0; r < 8; r++) {
                float p = __expf(sf[j][r] - newm[r]);
                sf[j][r] = p;
                tsum[r] += p;
            }
#pragma unroll
        for (int off = 1; off < 16; off <<= 1)
#pragma unroll
            for (int r = 0; r < 8; r++)
                tsum[r] += __shfl_xor(tsum[r], off, 16);
#pragma unroll
        for (int r = 0; r < 8; r++) {
            corr[r] = __expf(mrow[r] - newm[r]);
            lrow[r] = lrow[r] * corr[r] + tsum[r];
            mrow[r] = newm[r];
        }
#pragma unroll
        for (int j = 0; j < 4; j++)
#pragma unroll
            for (int r = 0; r < 8; r++) (*ofr[j])[r] *= corr[r];

        // C-layout -> A-layout via per-wave LDS round-trip
#pragma unroll
        for (int j = 0; j < 4; j++)
#pragma unroll
            for (int r = 0; r < 8; r++)
                Pw[(r + rowoff) * A_LS + j * 16 + (lane & 15)] = (_Float16)sf[j][r];

        v16h pa0 = load_frag_a(Pw, A_LS, 0, 0,  lane);
        v16h pa1 = load_frag_a(Pw, A_LS, 0, 32, lane);
        // O += P V   (4 dh-subtiles x 2 K-steps = 8 WMMAs)
#pragma unroll
        for (int j = 0; j < 4; j++) {
            v16h bv0 = load_frag_b(Vt, A_LS, j * 16, 0,  lane);
            v16h bv1 = load_frag_b(Vt, A_LS, j * 16, 32, lane);
            *ofr[j] = WMMA_F16(pa0, bv0, *ofr[j]);
            *ofr[j] = WMMA_F16(pa1, bv1, *ofr[j]);
        }
        __syncthreads();
    }

    float inv[8];
#pragma unroll
    for (int r = 0; r < 8; r++) inv[r] = 1.0f / lrow[r];
#pragma unroll
    for (int j = 0; j < 4; j++)
#pragma unroll
        for (int r = 0; r < 8; r++) {
            int m = q0 + wave * 16 + r + rowoff;
            int col = h * DH + j * 16 + (lane & 15);
            O[(size_t)(b * S + m) * D + col] = (_Float16)((*ofr[j])[r] * inv[r]);
        }
}

// ---------------------------------------------------------------------------
// Launch
// ---------------------------------------------------------------------------
extern "C" void kernel_launch(void* const* d_in, const int* in_sizes, int n_in,
                              void* d_out, int out_size, void* d_ws, size_t ws_size,
                              hipStream_t stream) {
    const int B = 2, S = 2048, D = 1024, H = 16;
    const int M = B * S;  // 4096

    const float* x    = (const float*)d_in[0];
    const int*   mask = (const int*)d_in[1];
    const float* wq   = (const float*)d_in[2];
    const float* bq   = (const float*)d_in[3];
    const float* wk   = (const float*)d_in[4];
    const float* bk   = (const float*)d_in[5];
    const float* wv   = (const float*)d_in[6];
    const float* bv   = (const float*)d_in[7];
    const float* wo   = (const float*)d_in[8];
    const float* bo   = (const float*)d_in[9];
    float* out = (float*)d_out;

    const size_t MiB = 1024 * 1024;
    char* ws = (char*)d_ws;
    _Float16* xh  = (_Float16*)(ws + 0 * MiB);    // 8 MiB  (4096x1024 f16)
    _Float16* wqT = (_Float16*)(ws + 8 * MiB);    // 2 MiB each
    _Float16* wkT = (_Float16*)(ws + 10 * MiB);
    _Float16* wvT = (_Float16*)(ws + 12 * MiB);
    _Float16* woT = (_Float16*)(ws + 14 * MiB);
    _Float16* Qh  = (_Float16*)(ws + 16 * MiB);   // 8 MiB each
    _Float16* Kh  = (_Float16*)(ws + 24 * MiB);
    _Float16* Vh  = (_Float16*)(ws + 32 * MiB);
    _Float16* Oh  = (_Float16*)(ws + 40 * MiB);

    // 1) convert x -> f16
    cvt_f32_to_f16<<<(M * D) / 1024, 256, 0, stream>>>(x, xh, M * D);

    // 2) transpose+convert weights -> [N][K] f16
    dim3 tb(32, 8), tg(D / 32, D / 32);
    transpose_cvt<<<tg, tb, 0, stream>>>(wq, wqT, D, D);
    transpose_cvt<<<tg, tb, 0, stream>>>(wk, wkT, D, D);
    transpose_cvt<<<tg, tb, 0, stream>>>(wv, wvT, D, D);
    transpose_cvt<<<tg, tb, 0, stream>>>(wo, woT, D, D);

    // 3) QKV projections (WMMA), f16 outputs in [B,S,D]
    dim3 gg(M / G_TM, D / G_TN);
    gemm_wmma_f16<<<gg, 256, 0, stream>>>(xh, wqT, bq, Qh, nullptr, M, D, D);
    gemm_wmma_f16<<<gg, 256, 0, stream>>>(xh, wkT, bk, Kh, nullptr, M, D, D);
    gemm_wmma_f16<<<gg, 256, 0, stream>>>(xh, wvT, bv, Vh, nullptr, M, D, D);

    // 4) RoPE on Q and K (in place)
    int rope_threads = B * S * H * 32;
    rope_kernel<<<rope_threads / 256, 256, 0, stream>>>(Qh, B, S, H, D);
    rope_kernel<<<rope_threads / 256, 256, 0, stream>>>(Kh, B, S, H, D);

    // 5) flash attention (WMMA) -> Oh f16 [B,S,D]
    dim3 ag(S / A_QT, H, B);
    attn_wmma<<<ag, 256, 0, stream>>>(Qh, Kh, Vh, mask, Oh, B, S, H);

    // 6) output projection (WMMA) -> f32 d_out
    gemm_wmma_f16<<<gg, 256, 0, stream>>>(Oh, woT, bo, nullptr, out, M, D, D);
}
